// SpatialAwareAttention_90486370992267
// MI455X (gfx1250) — matrix operand order repaired and checked by reading, not compiled, not run on hardware
//
#include <hip/hip_runtime.h>
#include <hip/hip_bf16.h>
#include <stdint.h>

// ---------------- CDNA5 WMMA types ----------------
typedef __attribute__((ext_vector_type(16))) __bf16 v16bf;
typedef __attribute__((ext_vector_type(8)))  float  v8f;

#define WMMA_BF16(a, b, c) \
  __builtin_amdgcn_wmma_f32_16x16x32_bf16(false, (a), false, (b), (short)0, (c), false, false)

// 16-byte chunk (avoid HIP vector-type ctor issues inside unions)
struct __attribute__((aligned(16))) U4 { unsigned int a, b, c, d; };
union Frag { U4 q[2]; v16bf v; };
struct F2 { float mu, rsig; };

// ---------------- problem constants ----------------
#define BB     4
#define NTOK   2049          // N = NPROTO + 1
#define DD     512
#define HH     8
#define HDIM   64
#define NC3    1536          // 3*D
#define NROW   (BB * NTOK)   // 8196
#define RPAD   8208          // 513 * 16
#define NQPAD  2064          // 129 * 16
#define NKPAD  2112          // 33 * 64

static __device__ __forceinline__ unsigned short f2bf(float f) {
  unsigned int u = __float_as_uint(f);
  u += 0x7FFFu + ((u >> 16) & 1u);     // round-to-nearest-even
  return (unsigned short)(u >> 16);
}

// ============ kernel 0: x -> bf16, per-row LayerNorm stats ============
__global__ void __launch_bounds__(128)
prep_kernel(const float* __restrict__ x, unsigned short* __restrict__ xbf,
            F2* __restrict__ stats) {
  int row = blockIdx.x;            // [0, RPAD)
  int tid = threadIdx.x;           // 128 threads
  __shared__ float red[8];
  if (row >= NROW) {               // zero-fill padded rows
    for (int i = 0; i < 4; ++i) xbf[(size_t)row * DD + tid + 128 * i] = 0;
    if (tid == 0) { stats[row].mu = 0.f; stats[row].rsig = 1.f; }
    return;
  }
  const float* xr = x + (size_t)row * DD;
  float v[4], s = 0.f, ss = 0.f;
  for (int i = 0; i < 4; ++i) {
    v[i] = xr[tid + 128 * i];
    s += v[i]; ss += v[i] * v[i];
  }
  for (int m = 16; m >= 1; m >>= 1) {
    s  += __shfl_xor(s,  m, 32);
    ss += __shfl_xor(ss, m, 32);
  }
  int wave = tid >> 5;
  if ((tid & 31) == 0) { red[wave] = s; red[4 + wave] = ss; }
  __syncthreads();
  s  = red[0] + red[1] + red[2] + red[3];
  ss = red[4] + red[5] + red[6] + red[7];
  float mu  = s * (1.0f / DD);
  float var = ss * (1.0f / DD) - mu * mu;
  float rsg = rsqrtf(var + 1e-5f);
  if (tid == 0) { stats[row].mu = mu; stats[row].rsig = rsg; }
  for (int i = 0; i < 4; ++i)
    xbf[(size_t)row * DD + tid + 128 * i] = f2bf(v[i]);
}

// ============ kernel 0b: W (K x Ncol, f32) -> W^T (Ncol x K, bf16) ============
__global__ void __launch_bounds__(256)
transpose_bf16_kernel(const float* __restrict__ w, unsigned short* __restrict__ wt,
                      int K, int Ncol) {
  int idx = blockIdx.x * 256 + threadIdx.x;
  if (idx >= K * Ncol) return;
  int k = idx / Ncol, n = idx - k * Ncol;
  wt[(size_t)n * K + k] = f2bf(w[(size_t)k * Ncol + n]);
}

// ============ kernel 1: QKV GEMM (bf16 WMMA), scatter Q/K/V^T ============
__global__ void __launch_bounds__(32)
qkv_gemm_kernel(const unsigned short* __restrict__ xbf,   // RPAD x DD
                const unsigned short* __restrict__ wT,    // NC3 x DD
                unsigned short* __restrict__ Qbf,         // (B*H) x NQPAD x HDIM
                unsigned short* __restrict__ Kbf,         // (B*H) x NKPAD x HDIM
                unsigned short* __restrict__ VT) {        // (B*H) x HDIM x NKPAD
  int lane = threadIdx.x, half = lane >> 4, nn = lane & 15;
  int rowbase = blockIdx.x * 16;
  int colbase = blockIdx.y * 64;
  v8f z = {0.f, 0.f, 0.f, 0.f, 0.f, 0.f, 0.f, 0.f};
  v8f acc[4] = {z, z, z, z};
  const unsigned short* arow = xbf + (size_t)(rowbase + nn) * DD;
  for (int kk = 0; kk < DD; kk += 32) {
    Frag a;
    a.q[0] = *(const U4*)(arow + kk + half * 16);
    a.q[1] = *(const U4*)(arow + kk + half * 16 + 8);
#pragma unroll
    for (int s4 = 0; s4 < 4; ++s4) {
      const unsigned short* brow =
          wT + (size_t)(colbase + s4 * 16 + nn) * DD + kk + half * 16;
      Frag b;
      b.q[0] = *(const U4*)(brow);
      b.q[1] = *(const U4*)(brow + 8);
      acc[s4] = WMMA_BF16(a.v, b.v, acc[s4]);
    }
  }
#pragma unroll
  for (int s4 = 0; s4 < 4; ++s4) {
    int j = colbase + s4 * 16 + nn;      // [0, 1536)
    int t = j / DD;  int rem = j - t * DD;
    int h = rem >> 6, hd = rem & 63;
#pragma unroll
    for (int r = 0; r < 8; ++r) {
      int rg = rowbase + r + 8 * half;
      if (rg >= NROW) continue;
      int b_ = rg / NTOK, n0 = rg - b_ * NTOK;
      int bh = b_ * HH + h;
      unsigned short val = f2bf(acc[s4][r]);
      if (t == 0)      Qbf[((size_t)bh * NQPAD + n0) * HDIM + hd] = val;
      else if (t == 1) Kbf[((size_t)bh * NKPAD + n0) * HDIM + hd] = val;
      else             VT[((size_t)bh * HDIM + hd) * NKPAD + n0]  = val;
    }
  }
}

// ============ kernel 2: flash attention with spatial gating ============
__global__ void __launch_bounds__(32)
attn_kernel(const unsigned short* __restrict__ Qbf,
            const unsigned short* __restrict__ Kbf,
            const unsigned short* __restrict__ VT,
            const float* __restrict__ sim,        // NTOK x NTOK
            const float* __restrict__ logit_ptr,  // scalar
            unsigned short* __restrict__ obf) {   // RPAD x DD (bf16)
  __shared__ __align__(16) unsigned short Pld[16][72];  // 16-row P stage, padded
  int lane = threadIdx.x, half = lane >> 4, nn = lane & 15;
  int qt = blockIdx.x;               // 129 query tiles
  int bh = blockIdx.y;               // 32 (b,h) pairs
  int b_ = bh >> 3, h_ = bh & 7;
  int qbase = qt * 16;
  float alpha = 1.0f / (1.0f + __expf(-logit_ptr[0]));
  float cs = (1.0f - alpha) * 0.125f;            // (1-alpha) * HD^-0.5

  // Q A-fragments (reused for all key blocks)
  Frag aq[2];
  const unsigned short* qrow = Qbf + ((size_t)bh * NQPAD + qbase + nn) * HDIM;
#pragma unroll
  for (int ka = 0; ka < 2; ++ka) {
    aq[ka].q[0] = *(const U4*)(qrow + ka * 32 + half * 16);
    aq[ka].q[1] = *(const U4*)(qrow + ka * 32 + half * 16 + 8);
  }

  v8f z = {0.f, 0.f, 0.f, 0.f, 0.f, 0.f, 0.f, 0.f};
  v8f o[4] = {z, z, z, z};
  float rm[8], rs[8];
#pragma unroll
  for (int r = 0; r < 8; ++r) { rm[r] = -3.0e38f; rs[r] = 0.f; }

  for (int kb = 0; kb < NTOK; kb += 64) {       // 33 key blocks
    float l[4][8];
#pragma unroll
    for (int j4 = 0; j4 < 4; ++j4) {
      int keyb = kb + j4 * 16;
      v8f s = z;
#pragma unroll
      for (int ka = 0; ka < 2; ++ka) {
        const unsigned short* kr =
            Kbf + ((size_t)bh * NKPAD + keyb + nn) * HDIM + ka * 32 + half * 16;
        Frag bk;
        bk.q[0] = *(const U4*)(kr);
        bk.q[1] = *(const U4*)(kr + 8);
        s = WMMA_BF16(aq[ka].v, bk.v, s);
      }
      int kcol = keyb + nn;
#pragma unroll
      for (int r = 0; r < 8; ++r) {
        int qr = qbase + r + 8 * half;
        if (qr > NTOK - 1) qr = NTOK - 1;       // clamp (masked at store)
        float lv = (kcol < NTOK)
                       ? cs * s[r] + alpha * sim[(size_t)qr * NTOK + kcol]
                       : -3.0e38f;
        l[j4][r] = lv;
      }
    }
    // online softmax per row (rows live on one 16-lane half)
#pragma unroll
    for (int r = 0; r < 8; ++r) {
      float mx = fmaxf(fmaxf(l[0][r], l[1][r]), fmaxf(l[2][r], l[3][r]));
      for (int m = 8; m >= 1; m >>= 1) mx = fmaxf(mx, __shfl_xor(mx, m, 32));
      float newm = fmaxf(rm[r], mx);
      float f = __expf(rm[r] - newm);
      float psum = 0.f;
#pragma unroll
      for (int j4 = 0; j4 < 4; ++j4) {
        float p = __expf(l[j4][r] - newm);
        l[j4][r] = p;
        psum += p;
      }
      for (int m = 8; m >= 1; m >>= 1) psum += __shfl_xor(psum, m, 32);
      rs[r] = rs[r] * f + psum;
      rm[r] = newm;
#pragma unroll
      for (int s4 = 0; s4 < 4; ++s4) o[s4][r] *= f;
    }
    // C-layout -> A-fragment layout via LDS
#pragma unroll
    for (int j4 = 0; j4 < 4; ++j4)
#pragma unroll
      for (int r = 0; r < 8; ++r)
        Pld[r + 8 * half][j4 * 16 + nn] = f2bf(l[j4][r]);
    __asm__ volatile("s_wait_dscnt 0" ::: "memory");
    Frag ap[2];
#pragma unroll
    for (int ka = 0; ka < 2; ++ka) {
      ap[ka].q[0] = *(const U4*)(&Pld[nn][ka * 32 + half * 16]);
      ap[ka].q[1] = *(const U4*)(&Pld[nn][ka * 32 + half * 16 + 8]);
    }
    // O += P @ V
#pragma unroll
    for (int s4 = 0; s4 < 4; ++s4) {
#pragma unroll
      for (int ka = 0; ka < 2; ++ka) {
        const unsigned short* vr =
            VT + ((size_t)bh * HDIM + s4 * 16 + nn) * NKPAD + kb + ka * 32 + half * 16;
        Frag bv;
        bv.q[0] = *(const U4*)(vr);
        bv.q[1] = *(const U4*)(vr + 8);
        o[s4] = WMMA_BF16(ap[ka].v, bv.v, o[s4]);
      }
    }
  }
  // normalize + store (B,N,D) bf16 for the proj GEMM
#pragma unroll
  for (int s4 = 0; s4 < 4; ++s4) {
#pragma unroll
    for (int r = 0; r < 8; ++r) {
      int q = qbase + r + 8 * half;
      if (q < NTOK) {
        float val = o[s4][r] / rs[r];
        obf[((size_t)(b_ * NTOK + q)) * DD + h_ * HDIM + s4 * 16 + nn] = f2bf(val);
      }
    }
  }
}

// ============ kernel 3: proj GEMM + bias + LayerNorm residual ============
__global__ void __launch_bounds__(32)
proj_gemm_kernel(const unsigned short* __restrict__ obf,   // RPAD x DD
                 const unsigned short* __restrict__ wT,    // DD x DD
                 const float* __restrict__ x,
                 const F2* __restrict__ stats,
                 const float* __restrict__ bproj,
                 const float* __restrict__ gamma,
                 const float* __restrict__ beta,
                 float* __restrict__ out) {
  int lane = threadIdx.x, half = lane >> 4, nn = lane & 15;
  int rowbase = blockIdx.x * 16;
  int colbase = blockIdx.y * 64;
  v8f z = {0.f, 0.f, 0.f, 0.f, 0.f, 0.f, 0.f, 0.f};
  v8f acc[4] = {z, z, z, z};
  const unsigned short* arow = obf + (size_t)(rowbase + nn) * DD;
  for (int kk = 0; kk < DD; kk += 32) {
    Frag a;
    a.q[0] = *(const U4*)(arow + kk + half * 16);
    a.q[1] = *(const U4*)(arow + kk + half * 16 + 8);
#pragma unroll
    for (int s4 = 0; s4 < 4; ++s4) {
      const unsigned short* brow =
          wT + (size_t)(colbase + s4 * 16 + nn) * DD + kk + half * 16;
      Frag b;
      b.q[0] = *(const U4*)(brow);
      b.q[1] = *(const U4*)(brow + 8);
      acc[s4] = WMMA_BF16(a.v, b.v, acc[s4]);
    }
  }
#pragma unroll
  for (int s4 = 0; s4 < 4; ++s4) {
    int j = colbase + s4 * 16 + nn;
#pragma unroll
    for (int r = 0; r < 8; ++r) {
      int rg = rowbase + r + 8 * half;
      if (rg < NROW) {
        F2 st = stats[rg];
        float ln = (x[(size_t)rg * DD + j] - st.mu) * st.rsig * gamma[j] + beta[j];
        out[(size_t)rg * DD + j] = acc[s4][r] + bproj[j] + ln;
      }
    }
  }
}

// ============ host launcher ============
extern "C" void kernel_launch(void* const* d_in, const int* in_sizes, int n_in,
                              void* d_out, int out_size, void* d_ws, size_t ws_size,
                              hipStream_t stream) {
  const float* x      = (const float*)d_in[0];
  const float* sim    = (const float*)d_in[1];
  const float* w_qkv  = (const float*)d_in[2];
  const float* w_proj = (const float*)d_in[3];
  const float* b_proj = (const float*)d_in[4];
  const float* gamma  = (const float*)d_in[5];
  const float* beta   = (const float*)d_in[6];
  const float* slogit = (const float*)d_in[7];

  char* ws = (char*)d_ws;
  size_t off = 0;
  auto carve = [&](size_t bytes) -> char* {
    char* p = ws + off;
    off += (bytes + 255) & ~(size_t)255;
    return p;
  };
  unsigned short* xbf    = (unsigned short*)carve((size_t)RPAD * DD * 2);
  F2*             stats  = (F2*)carve((size_t)RPAD * sizeof(F2));
  unsigned short* wqkvT  = (unsigned short*)carve((size_t)NC3 * DD * 2);
  unsigned short* wprojT = (unsigned short*)carve((size_t)DD * DD * 2);
  unsigned short* Qbf    = (unsigned short*)carve((size_t)BB * HH * NQPAD * HDIM * 2);
  unsigned short* Kbf    = (unsigned short*)carve((size_t)BB * HH * NKPAD * HDIM * 2);
  unsigned short* VT     = (unsigned short*)carve((size_t)BB * HH * HDIM * NKPAD * 2);
  unsigned short* obf    = (unsigned short*)carve((size_t)RPAD * DD * 2);

  prep_kernel<<<RPAD, 128, 0, stream>>>(x, xbf, stats);
  transpose_bf16_kernel<<<(DD * NC3 + 255) / 256, 256, 0, stream>>>(w_qkv, wqkvT, DD, NC3);
  transpose_bf16_kernel<<<(DD * DD + 255) / 256, 256, 0, stream>>>(w_proj, wprojT, DD, DD);
  qkv_gemm_kernel<<<dim3(RPAD / 16, NC3 / 64), 32, 0, stream>>>(xbf, wqkvT, Qbf, Kbf, VT);
  attn_kernel<<<dim3(NQPAD / 16, BB * HH), 32, 0, stream>>>(Qbf, Kbf, VT, sim, slogit, obf);
  proj_gemm_kernel<<<dim3(RPAD / 16, DD / 64), 32, 0, stream>>>(
      obf, wprojT, x, stats, b_proj, gamma, beta, (float*)d_out);
}
